// DualRelationsPropagation_17781164605533
// MI455X (gfx1250) — compile-verified, matrix operand order checked by reading, not run on hardware
//
#include <hip/hip_runtime.h>

typedef __attribute__((ext_vector_type(16))) __bf16 v16bf;
typedef __attribute__((ext_vector_type(8)))  __bf16 v8bf;
typedef __attribute__((ext_vector_type(8)))  float  v8f;

#define HDIM  768
#define NNODE 256
#define RDIM  16

// ---------------------------------------------------------------------------
// WMMA helpers (gfx1250 wave32, v_wmma_f32_16x16x32_bf16)
// ---------------------------------------------------------------------------
__device__ __forceinline__ v8f wmma_bf16(v16bf a, v16bf b, v8f c) {
  return __builtin_amdgcn_wmma_f32_16x16x32_bf16(
      /*neg_a=*/false, a, /*neg_b=*/false, b,
      /*c_mod=*/(short)0, c, /*reuse_a=*/false, /*reuse_b=*/false);
}

// abs via inline asm: opaque to InstCombine/SLP so the broken gfx1250
// vector-fabs DAG pattern (i32 bitcast of v2f32 + and 0x7fffffff) cannot form.
__device__ __forceinline__ float absf_safe(float d) {
  float r;
  asm volatile("v_and_b32 %0, 0x7fffffff, %1" : "=v"(r) : "v"(d));
  return r;
}

// A/B 16-bit fragment (16 rows x 32 K): lane L holds row (L&15); K elements are
// two contiguous 8-element runs: [kb..kb+7] and [16+kb..16+kb+7], kb = (L&16)?8:0.
// Works on LDS or global bf16 (row stride ld elements): two b128 loads.
__device__ __forceinline__ v16bf frag_lds(const __bf16* base, int row, int ld, int lane) {
  const int kb = (lane & 16) ? 8 : 0;
  const __bf16* p = base + (size_t)row * ld + kb;
  v8bf lo = *(const v8bf*)(p);
  v8bf hi = *(const v8bf*)(p + 16);
  v16bf f;
#pragma unroll
  for (int e = 0; e < 8; ++e) { f[e] = lo[e]; f[8 + e] = hi[e]; }
  return f;
}

// 16-lane-group reductions (softmax over the R=16 dim which sits across lanes)
__device__ __forceinline__ float rmax16(float v) {
  v = fmaxf(v, __shfl_xor(v, 1, 32));
  v = fmaxf(v, __shfl_xor(v, 2, 32));
  v = fmaxf(v, __shfl_xor(v, 4, 32));
  v = fmaxf(v, __shfl_xor(v, 8, 32));
  return v;
}
__device__ __forceinline__ float rsum16(float v) {
  v += __shfl_xor(v, 1, 32);
  v += __shfl_xor(v, 2, 32);
  v += __shfl_xor(v, 4, 32);
  v += __shfl_xor(v, 8, 32);
  return v;
}

// Softmax over N=16 (across lanes) for one 16x16 C/D tile, then store to
// out[(i*NNODE + j)*RDIM + r].
__device__ __forceinline__ void softmax_store(v8f acc, const float* b2,
                                              float* out, int i, int jbase, int lane) {
  const int r = lane & 15;
  const int joff = jbase + ((lane & 16) ? 8 : 0);
#pragma unroll
  for (int v = 0; v < 8; ++v) {
    float val = acc[v] + b2[r];
    float mx = rmax16(val);
    float e = __expf(val - mx);
    float s = rsum16(e);
    out[(size_t)((i * NNODE) + joff + v) * RDIM + r] = e / s;
  }
}

// ---------------------------------------------------------------------------
// Generic NT GEMM: Out[m,n] = act(alpha * sum_k A[m,k]*B[n,k] + bias[n] + res[m,n])
// Block = 256 threads (8 waves), 64x64 output tile, BK=32, bf16 WMMA.
// flags: bit0 = ReLU, bit1 = bf16 output.
// M = gridDim.y*64, N = gridDim.x*64 (all dims multiples of tile sizes).
// ---------------------------------------------------------------------------
__global__ __launch_bounds__(256) void gemm_nt_kernel(
    const float* __restrict__ A, int lda,
    const float* __restrict__ B, int ldb,
    const float* __restrict__ bias,
    const float* __restrict__ resid, int ldr,
    void* __restrict__ Out, int ldo,
    int K, float alpha, int flags) {
  __shared__ __bf16 sA[64 * 32];
  __shared__ __bf16 sB[64 * 32];
  const int t = threadIdx.x, lane = t & 31, wave = t >> 5;
  const int m0 = blockIdx.y * 64, n0 = blockIdx.x * 64;
  const int tr = wave >> 1;            // tile row 0..3
  const int tcb = (wave & 1) * 2;      // tile col base 0 or 2
  const int srow = t >> 2, skc = (t & 3) * 8;  // staging: 8 f32 per thread per matrix
  v8f acc0 = {}, acc1 = {};

  for (int kk = 0; kk < K; kk += 32) {
    __syncthreads();
    {
      const float4* pa = (const float4*)(A + (size_t)(m0 + srow) * lda + kk + skc);
      float4 a0 = pa[0], a1 = pa[1];
      v8bf va;
      va[0] = (__bf16)a0.x; va[1] = (__bf16)a0.y; va[2] = (__bf16)a0.z; va[3] = (__bf16)a0.w;
      va[4] = (__bf16)a1.x; va[5] = (__bf16)a1.y; va[6] = (__bf16)a1.z; va[7] = (__bf16)a1.w;
      *(v8bf*)&sA[srow * 32 + skc] = va;
      const float4* pb = (const float4*)(B + (size_t)(n0 + srow) * ldb + kk + skc);
      float4 b0 = pb[0], b1 = pb[1];
      v8bf vb;
      vb[0] = (__bf16)b0.x; vb[1] = (__bf16)b0.y; vb[2] = (__bf16)b0.z; vb[3] = (__bf16)b0.w;
      vb[4] = (__bf16)b1.x; vb[5] = (__bf16)b1.y; vb[6] = (__bf16)b1.z; vb[7] = (__bf16)b1.w;
      *(v8bf*)&sB[srow * 32 + skc] = vb;
    }
    __syncthreads();
    v16bf aF = frag_lds(sA, tr * 16 + (lane & 15), 32, lane);
    v16bf bF0 = frag_lds(sB, (tcb + 0) * 16 + (lane & 15), 32, lane);
    v16bf bF1 = frag_lds(sB, (tcb + 1) * 16 + (lane & 15), 32, lane);
    acc0 = wmma_bf16(aF, bF0, acc0);
    acc1 = wmma_bf16(aF, bF1, acc1);
  }

  const int mbase = m0 + tr * 16 + ((lane & 16) ? 8 : 0);
#pragma unroll
  for (int v = 0; v < 8; ++v) {
    const int m = mbase + v;
#pragma unroll
    for (int h = 0; h < 2; ++h) {
      const int n = n0 + (tcb + h) * 16 + (lane & 15);
      float val = (h ? acc1[v] : acc0[v]) * alpha;
      if (bias)  val += bias[n];
      if (resid) val += resid[(size_t)m * ldr + n];
      if (flags & 1) val = fmaxf(val, 0.0f);
      if (flags & 2) ((__bf16*)Out)[(size_t)m * ldo + n] = (__bf16)val;
      else           ((float*)Out)[(size_t)m * ldo + n] = val;
    }
  }
}

// ---------------------------------------------------------------------------
// NN GEMM for msg = (C @ T) * alpha : A f32 [M x K] row-major, B bf16 [K x N]
// row-major (K-major). Used for einsum('ijr,jrk->ik').
// ---------------------------------------------------------------------------
__global__ __launch_bounds__(256) void gemm_nn_kernel(
    const float* __restrict__ A, int lda,
    const __bf16* __restrict__ B, int ldb,
    float* __restrict__ Out, int ldo,
    int K, float alpha) {
  __shared__ __bf16 sA[64 * 32];
  __shared__ __bf16 sB[32 * 64];
  const int t = threadIdx.x, lane = t & 31, wave = t >> 5;
  const int m0 = blockIdx.y * 64, n0 = blockIdx.x * 64;
  const int tr = wave >> 1, tcb = (wave & 1) * 2;
  const int srow = t >> 2, skc = (t & 3) * 8;
  const int brow = t >> 3, bnc = (t & 7) * 8;
  v8f acc0 = {}, acc1 = {};

  for (int kk = 0; kk < K; kk += 32) {
    __syncthreads();
    {
      const float4* pa = (const float4*)(A + (size_t)(m0 + srow) * lda + kk + skc);
      float4 a0 = pa[0], a1 = pa[1];
      v8bf va;
      va[0] = (__bf16)a0.x; va[1] = (__bf16)a0.y; va[2] = (__bf16)a0.z; va[3] = (__bf16)a0.w;
      va[4] = (__bf16)a1.x; va[5] = (__bf16)a1.y; va[6] = (__bf16)a1.z; va[7] = (__bf16)a1.w;
      *(v8bf*)&sA[srow * 32 + skc] = va;
      *(v8bf*)&sB[brow * 64 + bnc] =
          *(const v8bf*)&B[(size_t)(kk + brow) * ldb + n0 + bnc];
    }
    __syncthreads();
    v16bf aF = frag_lds(sA, tr * 16 + (lane & 15), 32, lane);
    const int kb = (lane & 16) ? 8 : 0;
#pragma unroll
    for (int h = 0; h < 2; ++h) {
      const int n = (tcb + h) * 16 + (lane & 15);
      v16bf bF;
#pragma unroll
      for (int e = 0; e < 8; ++e) {
        bF[e]     = sB[(kb + e) * 64 + n];
        bF[8 + e] = sB[(16 + kb + e) * 64 + n];
      }
      if (h == 0) acc0 = wmma_bf16(aF, bF, acc0);
      else        acc1 = wmma_bf16(aF, bF, acc1);
    }
  }

  const int mbase = m0 + tr * 16 + ((lane & 16) ? 8 : 0);
#pragma unroll
  for (int v = 0; v < 8; ++v) {
    const int m = mbase + v;
    Out[(size_t)m * ldo + n0 + (tcb + 0) * 16 + (lane & 15)] = acc0[v] * alpha;
    Out[(size_t)m * ldo + n0 + (tcb + 1) * 16 + (lane & 15)] = acc1[v] * alpha;
  }
}

// ---------------------------------------------------------------------------
// f32 -> bf16 weight pre-conversion: dst[row*ncols+col] = src[row*lds_+coloff+col]
// ---------------------------------------------------------------------------
__global__ void cvt_bf16_kernel(const float* __restrict__ src, int lds_, int coloff,
                                __bf16* __restrict__ dst, int ncols) {
  const int row = blockIdx.y;
  const int col = blockIdx.x * 256 + threadIdx.x;
  if (col < ncols)
    dst[(size_t)row * ncols + col] = (__bf16)src[(size_t)row * lds_ + coloff + col];
}

// ---------------------------------------------------------------------------
// sim relation: for row i, 64 j's: hid = relu(pa[i]+pb[j]+b1) (LDS bf16 tile),
// logits = hid @ w2.T + b2 (WMMA, N=16), softmax over R, store raw (unsymmetrized).
// grid (4 j-chunks, 256 i), 128 threads (4 waves, one 16-j tile each).
// ---------------------------------------------------------------------------
__global__ __launch_bounds__(128) void relpair_sim_kernel(
    const float* __restrict__ pa, const float* __restrict__ pb,
    const float* __restrict__ b1, const __bf16* __restrict__ w2bf,
    const float* __restrict__ b2, float* __restrict__ out) {
  __shared__ __bf16 sA[64 * 32];
  const int i = blockIdx.y, j0 = blockIdx.x * 64;
  const int t = threadIdx.x, lane = t & 31, wave = t >> 5;
  const int srow = t >> 1, skc = (t & 1) * 16;
  v8f acc = {};

  for (int kk = 0; kk < HDIM; kk += 32) {
    __syncthreads();
    {
      const float* ppa = pa + (size_t)i * HDIM + kk + skc;
      const float* ppb = pb + (size_t)(j0 + srow) * HDIM + kk + skc;
      const float* pb1 = b1 + kk + skc;
#pragma unroll
      for (int e = 0; e < 16; ++e) {
        float v = ppa[e] + ppb[e] + pb1[e];
        sA[srow * 32 + skc + e] = (__bf16)fmaxf(v, 0.0f);
      }
    }
    __syncthreads();
    v16bf aF = frag_lds(sA, wave * 16 + (lane & 15), 32, lane);
    v16bf bF = frag_lds(w2bf + kk, lane & 15, HDIM, lane);
    acc = wmma_bf16(aF, bF, acc);
  }
  softmax_store(acc, b2, out, i, j0 + wave * 16, lane);
}

// ---------------------------------------------------------------------------
// div relation (dominant kernel): per (i, 64 j's):
//  phase0: D = |x_i - x_j|  -> LDS bf16 (64x768)
//  phase1: Hid = relu(pa_div[i] + D @ wb.T + b1) -> LDS bf16 (64x768), WMMA
//          (wb pre-converted to bf16: B frags are 2x global_load_b128, no cvt)
//  phase2: logits = Hid @ w2.T + b2, softmax -> raw out
// 256 threads, 192KB dynamic LDS.
// ---------------------------------------------------------------------------
__global__ __launch_bounds__(256) void div_pairs_kernel(
    const float* __restrict__ x, const float* __restrict__ pa,
    const float* __restrict__ b1, const __bf16* __restrict__ wbbf,
    const __bf16* __restrict__ w2bf, const float* __restrict__ b2,
    float* __restrict__ out) {
  extern __shared__ __bf16 dsm[];
  __bf16* sD = dsm;               // 64*768
  __bf16* sH = dsm + 64 * HDIM;   // 64*768
  const int i = blockIdx.y, j0 = blockIdx.x * 64;
  const int t = threadIdx.x, lane = t & 31, wave = t >> 5;

  // phase 0: |x_i - x_j| tile, f32 -> bf16 (abs via opaque v_and_b32)
  for (int b = t * 8; b < 64 * HDIM; b += 256 * 8) {
    const int j = b / HDIM, k = b % HDIM;
    const float4* qi = (const float4*)(x + (size_t)i * HDIM + k);
    const float4* qj = (const float4*)(x + (size_t)(j0 + j) * HDIM + k);
    float4 i0 = qi[0], i1 = qi[1], jA = qj[0], jB = qj[1];
    v8bf v;
    v[0] = (__bf16)absf_safe(i0.x - jA.x); v[1] = (__bf16)absf_safe(i0.y - jA.y);
    v[2] = (__bf16)absf_safe(i0.z - jA.z); v[3] = (__bf16)absf_safe(i0.w - jA.w);
    v[4] = (__bf16)absf_safe(i1.x - jB.x); v[5] = (__bf16)absf_safe(i1.y - jB.y);
    v[6] = (__bf16)absf_safe(i1.z - jB.z); v[7] = (__bf16)absf_safe(i1.w - jB.w);
    *(v8bf*)&sD[b] = v;
  }
  __syncthreads();

  // phase 1: Hid = relu(pa_i + D @ wb.T + b1)
  const int tr = wave >> 1, tcb = (wave & 1) * 2;
  for (int oc = 0; oc < 12; ++oc) {
    v8f acc0 = {}, acc1 = {};
    const int nA = oc * 64 + (tcb + 0) * 16 + (lane & 15);
    const int nB = oc * 64 + (tcb + 1) * 16 + (lane & 15);
    for (int kk = 0; kk < HDIM; kk += 32) {
      v16bf aF = frag_lds(sD + kk, tr * 16 + (lane & 15), HDIM, lane);
      __builtin_prefetch(wbbf + (size_t)nA * HDIM + kk + 32, 0, 3);
      v16bf bF0 = frag_lds(wbbf + kk, nA, HDIM, lane);
      v16bf bF1 = frag_lds(wbbf + kk, nB, HDIM, lane);
      acc0 = wmma_bf16(aF, bF0, acc0);
      acc1 = wmma_bf16(aF, bF1, acc1);
    }
#pragma unroll
    for (int v = 0; v < 8; ++v) {
      const int j = tr * 16 + v + ((lane & 16) ? 8 : 0);
      float val0 = acc0[v] + pa[(size_t)i * HDIM + nA] + b1[nA];
      sH[(size_t)j * HDIM + nA] = (__bf16)fmaxf(val0, 0.0f);
      float val1 = acc1[v] + pa[(size_t)i * HDIM + nB] + b1[nB];
      sH[(size_t)j * HDIM + nB] = (__bf16)fmaxf(val1, 0.0f);
    }
  }
  __syncthreads();

  // phase 2: logits = Hid @ w2.T + b2, softmax over R=16
  if (wave < 4) {
    v8f acc = {};
    for (int kk = 0; kk < HDIM; kk += 32) {
      v16bf aF = frag_lds(sH + kk, wave * 16 + (lane & 15), HDIM, lane);
      v16bf bF = frag_lds(w2bf + kk, lane & 15, HDIM, lane);
      acc = wmma_bf16(aF, bF, acc);
    }
    softmax_store(acc, b2, out, i, j0 + wave * 16, lane);
  }
}

// ---------------------------------------------------------------------------
// Symmetrize sim/div, zero diagonal, write outputs and C = 0.5*(sim+div)
// ---------------------------------------------------------------------------
__global__ __launch_bounds__(256) void sym_combine_kernel(
    const float* __restrict__ sraw, const float* __restrict__ draw,
    float* __restrict__ osim, float* __restrict__ odiv, float* __restrict__ C) {
  const int idx = blockIdx.x * 256 + threadIdx.x;  // over 256*256*16
  const int r = idx & 15, j = (idx >> 4) & 255, i = idx >> 12;
  float s, d;
  if (i == j) { s = 0.0f; d = 0.0f; }
  else if (i > j) {
    const int sw = ((j * NNODE + i) * RDIM) + r;
    s = sraw[sw]; d = draw[sw];
  } else { s = sraw[idx]; d = draw[idx]; }
  osim[idx] = s; odiv[idx] = d; C[idx] = 0.5f * (s + d);
}

// ---------------------------------------------------------------------------
// LayerNorm over H=768: x = (y-mu)/sqrt(var+1e-5)*g + b. One block per row.
// ---------------------------------------------------------------------------
__global__ __launch_bounds__(256) void ln_kernel(
    const float* __restrict__ y, const float* __restrict__ g,
    const float* __restrict__ b, float* __restrict__ outx) {
  __shared__ float red[256];
  const int row = blockIdx.x, t = threadIdx.x;
  const float v0 = y[(size_t)row * HDIM + t];
  const float v1 = y[(size_t)row * HDIM + 256 + t];
  const float v2 = y[(size_t)row * HDIM + 512 + t];
  red[t] = v0 + v1 + v2;
  __syncthreads();
  for (int off = 128; off > 0; off >>= 1) {
    if (t < off) red[t] += red[t + off];
    __syncthreads();
  }
  const float mu = red[0] / (float)HDIM;
  __syncthreads();
  const float d0 = v0 - mu, d1 = v1 - mu, d2 = v2 - mu;
  red[t] = d0 * d0 + d1 * d1 + d2 * d2;
  __syncthreads();
  for (int off = 128; off > 0; off >>= 1) {
    if (t < off) red[t] += red[t + off];
    __syncthreads();
  }
  const float rs = rsqrtf(red[0] / (float)HDIM + 1e-5f);
  outx[(size_t)row * HDIM + t]       = d0 * rs * g[t]       + b[t];
  outx[(size_t)row * HDIM + 256 + t] = d1 * rs * g[256 + t] + b[256 + t];
  outx[(size_t)row * HDIM + 512 + t] = d2 * rs * g[512 + t] + b[512 + t];
}

// ---------------------------------------------------------------------------
// strided 2D copy (rows = gridDim.y)
// ---------------------------------------------------------------------------
__global__ void copy2d_kernel(const float* __restrict__ src, int lds_,
                              float* __restrict__ dst, int ldd, int ncols) {
  const int row = blockIdx.y;
  const int col = blockIdx.x * 256 + threadIdx.x;
  if (col < ncols) dst[(size_t)row * ldd + col] = src[(size_t)row * lds_ + col];
}

// ---------------------------------------------------------------------------
// tiny classifier head: preds = clsh(128x384) @ cls_w2.T(3x384) + b2
// ---------------------------------------------------------------------------
__global__ void preds_kernel(const float* __restrict__ h,
                             const float* __restrict__ w2,
                             const float* __restrict__ b2,
                             float* __restrict__ out) {
  const int m = blockIdx.x * 64 + threadIdx.x;
  if (m >= 128) return;
#pragma unroll
  for (int c = 0; c < 3; ++c) {
    float s = b2[c];
    for (int k = 0; k < 384; ++k) s += h[(size_t)m * 384 + k] * w2[c * 384 + k];
    out[m * 3 + c] = s;
  }
}

// ---------------------------------------------------------------------------
extern "C" void kernel_launch(void* const* d_in, const int* in_sizes, int n_in,
                              void* d_out, int out_size, void* d_ws, size_t ws_size,
                              hipStream_t stream) {
  (void)in_sizes; (void)n_in; (void)out_size; (void)ws_size;
  const float* support = (const float*)d_in[0];
  const float* query   = (const float*)d_in[2];
  const float* sim_w1  = (const float*)d_in[3];
  const float* sim_b1  = (const float*)d_in[4];
  const float* sim_w2  = (const float*)d_in[5];
  const float* sim_b2  = (const float*)d_in[6];
  const float* div_w1  = (const float*)d_in[7];
  const float* div_b1  = (const float*)d_in[8];
  const float* div_w2  = (const float*)d_in[9];
  const float* div_b2  = (const float*)d_in[10];
  const float* rel_w   = (const float*)d_in[11];
  const float* rel_b   = (const float*)d_in[12];
  const float* agg_w1  = (const float*)d_in[13];
  const float* agg_b1  = (const float*)d_in[14];
  const float* agg_w2  = (const float*)d_in[15];
  const float* agg_b2  = (const float*)d_in[16];
  const float* ln_g    = (const float*)d_in[17];
  const float* ln_b    = (const float*)d_in[18];
  const float* cls_w1  = (const float*)d_in[19];
  const float* cls_b1  = (const float*)d_in[20];
  const float* cls_w2  = (const float*)d_in[21];
  const float* cls_b2  = (const float*)d_in[22];

  float* out = (float*)d_out;
  float* o_preds = out;
  float* o_sim   = out + 384;
  float* o_div   = o_sim + (size_t)NNODE * NNODE * RDIM;
  float* o_x     = o_div + (size_t)NNODE * NNODE * RDIM;

  float* W = (float*)d_ws;
  size_t off = 0;
  auto alloc = [&](size_t n) { float* p = W + off; off += n; return p; };
  float*  xbuf   = alloc((size_t)NNODE * HDIM);
  float*  pa_s   = alloc((size_t)NNODE * HDIM);
  float*  pb_s   = alloc((size_t)NNODE * HDIM);
  float*  pa_d   = alloc((size_t)NNODE * HDIM);
  float*  sraw   = alloc((size_t)NNODE * NNODE * RDIM);
  float*  draw   = alloc((size_t)NNODE * NNODE * RDIM);
  float*  Cbuf   = alloc((size_t)NNODE * NNODE * RDIM);
  __bf16* Tbf    = (__bf16*)alloc((size_t)NNODE * RDIM * HDIM / 2);  // bf16 [256][16][768]
  float*  comb   = alloc((size_t)NNODE * 2 * HDIM);
  float*  hidden = alloc((size_t)NNODE * HDIM);
  float*  aggpre = alloc((size_t)NNODE * HDIM);
  float*  clsh   = alloc((size_t)128 * 384);
  __bf16* wb_bf  = (__bf16*)alloc((size_t)HDIM * HDIM / 2);   // div_w1[:,H:] bf16
  __bf16* w2d_bf = (__bf16*)alloc((size_t)RDIM * HDIM / 2);   // div_w2 bf16
  __bf16* w2s_bf = (__bf16*)alloc((size_t)RDIM * HDIM / 2);   // sim_w2 bf16

  // all_x = concat(support, query)
  copy2d_kernel<<<dim3(3, 128), 256, 0, stream>>>(support, HDIM, xbuf, HDIM, HDIM);
  copy2d_kernel<<<dim3(3, 128), 256, 0, stream>>>(query, HDIM, xbuf + 128 * HDIM, HDIM, HDIM);

  // pre-convert hot-loop weights to bf16 (reused by 1024 blocks each)
  cvt_bf16_kernel<<<dim3(3, HDIM), 256, 0, stream>>>(div_w1, 2 * HDIM, HDIM, wb_bf, HDIM);
  cvt_bf16_kernel<<<dim3(3, RDIM), 256, 0, stream>>>(div_w2, HDIM, 0, w2d_bf, HDIM);
  cvt_bf16_kernel<<<dim3(3, RDIM), 256, 0, stream>>>(sim_w2, HDIM, 0, w2s_bf, HDIM);

  // pa_sim = x@wa.T, pb_sim = x@wb.T, pa_div = x@wa_div.T
  gemm_nt_kernel<<<dim3(12, 4), 256, 0, stream>>>(xbuf, HDIM, sim_w1, 2 * HDIM,
      nullptr, nullptr, 0, pa_s, HDIM, HDIM, 1.0f, 0);
  gemm_nt_kernel<<<dim3(12, 4), 256, 0, stream>>>(xbuf, HDIM, sim_w1 + HDIM, 2 * HDIM,
      nullptr, nullptr, 0, pb_s, HDIM, HDIM, 1.0f, 0);
  gemm_nt_kernel<<<dim3(12, 4), 256, 0, stream>>>(xbuf, HDIM, div_w1, 2 * HDIM,
      nullptr, nullptr, 0, pa_d, HDIM, HDIM, 1.0f, 0);

  relpair_sim_kernel<<<dim3(4, NNODE), 128, 0, stream>>>(pa_s, pb_s, sim_b1, w2s_bf, sim_b2, sraw);
  div_pairs_kernel<<<dim3(4, NNODE), 256, 2 * 64 * HDIM * sizeof(__bf16), stream>>>(
      xbuf, pa_d, div_b1, wb_bf, w2d_bf, div_b2, draw);
  sym_combine_kernel<<<4096, 256, 0, stream>>>(sraw, draw, o_sim, o_div, Cbuf);

  for (int l = 0; l < 3; ++l) {
    // T[j,r,k] = x @ rel_w[l,r].T + rel_b[l,r]  (stored bf16, element [j][r][k])
    for (int r = 0; r < 16; ++r) {
      gemm_nt_kernel<<<dim3(12, 4), 256, 0, stream>>>(
          xbuf, HDIM,
          rel_w + (size_t)(l * 16 + r) * HDIM * HDIM, HDIM,
          rel_b + (size_t)(l * 16 + r) * HDIM, nullptr, 0,
          (void*)(Tbf + (size_t)r * HDIM), RDIM * HDIM, HDIM, 1.0f, 2);
    }
    // comb = [x | msg], msg = (C @ T)/255 via NN GEMM (C: 256x4096, T: 4096x768)
    copy2d_kernel<<<dim3(3, NNODE), 256, 0, stream>>>(xbuf, HDIM, comb, 2 * HDIM, HDIM);
    gemm_nn_kernel<<<dim3(12, 4), 256, 0, stream>>>(Cbuf, NNODE * RDIM, Tbf, HDIM,
        comb + HDIM, 2 * HDIM, NNODE * RDIM, 1.0f / 255.0f);
    // agg MLP + residual
    gemm_nt_kernel<<<dim3(12, 4), 256, 0, stream>>>(comb, 2 * HDIM,
        agg_w1 + (size_t)l * HDIM * 2 * HDIM, 2 * HDIM,
        agg_b1 + (size_t)l * HDIM, nullptr, 0, hidden, HDIM, 2 * HDIM, 1.0f, 1);
    gemm_nt_kernel<<<dim3(12, 4), 256, 0, stream>>>(hidden, HDIM,
        agg_w2 + (size_t)l * HDIM * HDIM, HDIM,
        agg_b2 + (size_t)l * HDIM, xbuf, HDIM, aggpre, HDIM, HDIM, 1.0f, 0);
    ln_kernel<<<NNODE, 256, 0, stream>>>(aggpre, ln_g + (size_t)l * HDIM,
                                         ln_b + (size_t)l * HDIM, xbuf);
  }

  // classifier head on query rows
  gemm_nt_kernel<<<dim3(6, 2), 256, 0, stream>>>(xbuf + (size_t)128 * HDIM, HDIM,
      cls_w1, HDIM, cls_b1, nullptr, 0, clsh, 384, HDIM, 1.0f, 1);
  preds_kernel<<<2, 64, 0, stream>>>(clsh, cls_w2, cls_b2, o_preds);

  // final node embeddings
  copy2d_kernel<<<dim3(3, NNODE), 256, 0, stream>>>(xbuf, HDIM, o_x, HDIM, HDIM);
}
